// SoftClusterAttentionLayer_75952201662470
// MI455X (gfx1250) — matrix operand order repaired and checked by reading, compiled, not run on hardware
//
#include <hip/hip_runtime.h>
#include <hip/hip_bf16.h>
#include <math.h>

#define L_SEQ 1024
#define BATCH 4
#define EMB   1024
#define NHEAD 16
#define HDIM  64
#define BH    (BATCH * NHEAD)   // 64
#define MROWS (L_SEQ * BATCH)   // 4096

typedef __attribute__((ext_vector_type(16))) _Float16 v16h;
typedef __attribute__((ext_vector_type(8)))  _Float16 v8h;
typedef __attribute__((ext_vector_type(4)))  _Float16 v4h;
typedef __attribute__((ext_vector_type(8)))  float    v8f;

union AFrag { v16h v; v8h h[2]; };

// gfx1250 async global->LDS copy path (ASYNCcnt-tracked, no VGPR round trip).
// Probe-learned signature: (int4 addrspace(1)*, int4 addrspace(3)*, Ii, Ii).
#if defined(__AMDGCN__) && __has_builtin(__builtin_amdgcn_global_load_async_to_lds_b128)
#define USE_ASYNC_LDS 1
typedef int v4i_vec __attribute__((vector_size(16)));
#define GLB_V4I(p) ((__attribute__((address_space(1))) v4i_vec*)(p))
#define LDS_V4I(p) ((__attribute__((address_space(3))) v4i_vec*)(p))
#else
#define USE_ASYNC_LDS 0
#endif

__device__ __forceinline__ void async_stage_wait()
{
#if USE_ASYNC_LDS
#if __has_builtin(__builtin_amdgcn_s_wait_asynccnt)
  __builtin_amdgcn_s_wait_asynccnt(0);
#else
  asm volatile("s_wait_asynccnt 0" ::: "memory");
#endif
#endif
}

// Stage one 64-row x 32-half chunk of Brow ([N x K] row-major) into LDS.
// 256 threads x 16B each = 4 KB tile.
__device__ __forceinline__ void stage_b_chunk(const _Float16* __restrict__ Bblk,
                                              int ldb, int k0,
                                              _Float16* __restrict__ dst)
{
  const int tid = threadIdx.x;
  const int row = tid >> 2;
  const int cg  = (tid & 3) * 8;
  const _Float16* src = Bblk + (size_t)row * ldb + k0 + cg;
  _Float16* d = dst + row * 32 + cg;
#if USE_ASYNC_LDS
  __builtin_amdgcn_global_load_async_to_lds_b128(GLB_V4I(src), LDS_V4I(d), 0, 0);
#else
  *(v8h*)d = *(const v8h*)src;
#endif
}

// ---------------------------------------------------------------------------
// Shared GEMM core: workgroup = 256 threads (8 waves), computes a 128(M)x64(N)
// tile of C = A * Brow^T, A [M x K] row-major (lda), Brow [N x K] row-major
// (ldb).  Double-buffered LDS B tile: chunk k+1 is staged asynchronously
// (global_load_async_to_lds_b128) while 4x v_wmma_f32_16x16x32_f16 consume
// chunk k; one s_wait_asynccnt + one workgroup barrier per K chunk.
// lds_b must hold 2 * 64 * 32 halfs (8 KB).
// ---------------------------------------------------------------------------
__device__ __forceinline__ void gemm_core_128x64(
    const _Float16* __restrict__ Ablk, int lda,
    const _Float16* __restrict__ Bblk, int ldb,
    int K, _Float16* lds_b, v8f acc[4])
{
  const int tid    = threadIdx.x;
  const int lane   = tid & 31;
  const int wave   = tid >> 5;
  const int half16 = lane >> 4;     // 0: lanes 0-15, 1: lanes 16-31
  const int l16    = lane & 15;
  const _Float16* __restrict__ arow = Ablk + (size_t)(wave * 16 + l16) * lda;

  // prologue: stage chunk 0 into buffer 0
  stage_b_chunk(Bblk, ldb, 0, lds_b);
  async_stage_wait();
  __syncthreads();

  int buf = 0;
  for (int k0 = 0; k0 < K; k0 += 32) {
    // stage next chunk into the other buffer while we compute on this one
    if (k0 + 32 < K) {
      stage_b_chunk(Bblk, ldb, k0 + 32, lds_b + ((buf ^ 1) * (64 * 32)));
      __builtin_prefetch(arow + k0 + 32, 0, 2);   // global_prefetch_b8 for A
    }

    // A fragment: 16x32 f16, contiguous K order (matches B from LDS)
    // lanes 0-15 : halfs[0..15]=K(k0..k0+15)
    // lanes 16-31: halfs[0..15]=K(k0+16..k0+31)
    AFrag a;
    a.h[0] = *(const v8h*)(arow + k0 + half16 * 16);
    a.h[1] = *(const v8h*)(arow + k0 + half16 * 16 + 8);

    const _Float16* cur = lds_b + buf * (64 * 32);
#pragma unroll
    for (int t = 0; t < 4; ++t) {
      const v16h b = *(const v16h*)(cur + (t * 16 + l16) * 32 + half16 * 16);
      acc[t] = __builtin_amdgcn_wmma_f32_16x16x32_f16(
          false, a.v, false, b, (short)0, acc[t], false, false);
    }

    async_stage_wait();   // our async copies for chunk k+1 have landed
    __syncthreads();      // everyone done reading buf / writing buf^1
    buf ^= 1;
  }
}

// ---------------------------------------------------------------------------
// f32 -> f16 conversion, 4 elements/thread
// ---------------------------------------------------------------------------
__global__ void cvt4_f32_f16_kernel(const float* __restrict__ in,
                                    _Float16* __restrict__ out, int n4)
{
  const int i = blockIdx.x * blockDim.x + threadIdx.x;
  if (i < n4) {
    const float4 f = ((const float4*)in)[i];
    v4h h;
    h.x = (_Float16)f.x; h.y = (_Float16)f.y;
    h.z = (_Float16)f.z; h.w = (_Float16)f.w;
    ((v4h*)out)[i] = h;
  }
}

// ---------------------------------------------------------------------------
// QKV projection: x[z] (4096x1024) @ W[z]^T + b[z]; z in {q,k,v}.
// q -> head-major [bh][l][d], pre-scaled by 1/sqrt(64)
// k -> head-major [bh][s][d]
// v -> transposed  [bh][d][s]  (so AV GEMM's B operand is contiguous in K)
// ---------------------------------------------------------------------------
__global__ void qkv_proj_kernel(const _Float16* __restrict__ x16,
                                const _Float16* __restrict__ w_in16,
                                const float* __restrict__ bias_in,
                                _Float16* __restrict__ qh,
                                _Float16* __restrict__ kh,
                                _Float16* __restrict__ vt)
{
  __shared__ _Float16 lds_b[2 * 64 * 32];
  const int z      = blockIdx.z;
  const int blockM = blockIdx.x * 128;
  const int blockN = blockIdx.y * 64;
  const _Float16* A  = x16    + (size_t)z * MROWS * EMB + (size_t)blockM * EMB;
  const _Float16* Bp = w_in16 + (size_t)z * EMB * EMB   + (size_t)blockN * EMB;

  v8f acc[4] = {};
  gemm_core_128x64(A, EMB, Bp, EMB, EMB, lds_b, acc);

  const int lane = threadIdx.x & 31, wave = threadIdx.x >> 5;
  const int half16 = lane >> 4, l16 = lane & 15;
  const float scale = (z == 0) ? 0.125f : 1.0f;  // 1/sqrt(hd) for q
#pragma unroll
  for (int t = 0; t < 4; ++t) {
    const int n = blockN + t * 16 + l16;
    const int h = n >> 6, d = n & 63;
    const float bv = bias_in[z * EMB + n];
#pragma unroll
    for (int g = 0; g < 8; ++g) {
      const int m  = blockM + wave * 16 + g + half16 * 8;
      const int l  = m >> 2, b = m & 3;       // row m = l*B + b
      const int bh = b * NHEAD + h;
      const float val = (acc[t][g] + bv) * scale;
      if (z == 0)
        qh[(size_t)bh * L_SEQ * HDIM + (size_t)l * HDIM + d] = (_Float16)val;
      else if (z == 1)
        kh[(size_t)bh * L_SEQ * HDIM + (size_t)l * HDIM + d] = (_Float16)val;
      else
        vt[(size_t)bh * HDIM * L_SEQ + (size_t)d * L_SEQ + l] = (_Float16)val;
    }
  }
}

// ---------------------------------------------------------------------------
// Scores: S[bh] = q[bh] (1024x64) @ k[bh]^T -> f16 [bh][t][s]
// ---------------------------------------------------------------------------
__global__ void scores_kernel(const _Float16* __restrict__ qh,
                              const _Float16* __restrict__ kh,
                              _Float16* __restrict__ attn)
{
  __shared__ _Float16 lds_b[2 * 64 * 32];
  const int bh     = blockIdx.z;
  const int blockM = blockIdx.x * 128;   // t
  const int blockN = blockIdx.y * 64;    // s
  const _Float16* A  = qh + (size_t)bh * L_SEQ * HDIM + (size_t)blockM * HDIM;
  const _Float16* Bp = kh + (size_t)bh * L_SEQ * HDIM + (size_t)blockN * HDIM;

  v8f acc[4] = {};
  gemm_core_128x64(A, HDIM, Bp, HDIM, HDIM, lds_b, acc);

  const int lane = threadIdx.x & 31, wave = threadIdx.x >> 5;
  const int half16 = lane >> 4, l16 = lane & 15;
  _Float16* out = attn + (size_t)bh * L_SEQ * L_SEQ;
#pragma unroll
  for (int t = 0; t < 4; ++t) {
    const int n = blockN + t * 16 + l16;
#pragma unroll
    for (int g = 0; g < 8; ++g) {
      const int m = blockM + wave * 16 + g + half16 * 8;
      out[(size_t)m * L_SEQ + n] = (_Float16)acc[t][g];
    }
  }
}

// ---------------------------------------------------------------------------
// Softmax over the QUERY axis t, per (bh, s) column, in place on f16.
// (The reference's subsequent attn / attn.sum(axis=1) is the identity.)
// Online max+sum (single read pass) + normalize pass: 2 sweeps over 128 MB
// instead of 3.  Block: 256 threads = 4 t-groups x 64 s columns.
// ---------------------------------------------------------------------------
__global__ void softmax_t_kernel(_Float16* __restrict__ attn)
{
  const int bh = blockIdx.y;
  const int s0 = blockIdx.x * 64;
  _Float16* base = attn + (size_t)bh * L_SEQ * L_SEQ + s0;
  const int sx = threadIdx.x & 63;
  const int ty = threadIdx.x >> 6;
  __shared__ float redm[4 * 64];
  __shared__ float reds[4 * 64];

  float m = -3.0e38f, s = 0.f;
  for (int t = ty; t < L_SEQ; t += 4) {
    const float x  = (float)base[(size_t)t * L_SEQ + sx];
    const float nm = fmaxf(m, x);
    s = s * __expf(m - nm) + __expf(x - nm);
    m = nm;
  }
  redm[ty * 64 + sx] = m;
  reds[ty * 64 + sx] = s;
  __syncthreads();
  const float M = fmaxf(fmaxf(redm[sx], redm[64 + sx]),
                        fmaxf(redm[128 + sx], redm[192 + sx]));
  const float S = reds[sx]        * __expf(redm[sx]        - M)
                + reds[64 + sx]   * __expf(redm[64 + sx]   - M)
                + reds[128 + sx]  * __expf(redm[128 + sx]  - M)
                + reds[192 + sx]  * __expf(redm[192 + sx]  - M);
  const float inv = 1.0f / S;

  for (int t = ty; t < L_SEQ; t += 4) {
    const size_t o = (size_t)t * L_SEQ + sx;
    base[o] = (_Float16)(__expf((float)base[o] - M) * inv);
  }
}

// ---------------------------------------------------------------------------
// AV: out[bh] = attn[bh] (1024x1024) @ v[bh] (1024x64), scattered directly
// into the [L,B,E]-flat f16 layout for the out-projection.
// ---------------------------------------------------------------------------
__global__ void av_kernel(const _Float16* __restrict__ attn,
                          const _Float16* __restrict__ vt,
                          _Float16* __restrict__ oflat)
{
  __shared__ _Float16 lds_b[2 * 64 * 32];
  const int bh     = blockIdx.z;
  const int blockM = blockIdx.x * 128;   // t
  const _Float16* A  = attn + (size_t)bh * L_SEQ * L_SEQ + (size_t)blockM * L_SEQ;
  const _Float16* Bp = vt + (size_t)bh * HDIM * L_SEQ;  // rows d, ldb = L

  v8f acc[4] = {};
  gemm_core_128x64(A, L_SEQ, Bp, L_SEQ, L_SEQ, lds_b, acc);

  const int lane = threadIdx.x & 31, wave = threadIdx.x >> 5;
  const int half16 = lane >> 4, l16 = lane & 15;
  const int b = bh >> 4, h = bh & 15;
#pragma unroll
  for (int t = 0; t < 4; ++t) {
    const int d = t * 16 + l16;
#pragma unroll
    for (int g = 0; g < 8; ++g) {
      const int l = blockM + wave * 16 + g + half16 * 8;
      oflat[(size_t)(l * BATCH + b) * EMB + h * HDIM + d] = (_Float16)acc[t][g];
    }
  }
}

// ---------------------------------------------------------------------------
// Out projection: oflat (4096x1024) @ Wout^T + bout -> f32 d_out [L,B,E]
// ---------------------------------------------------------------------------
__global__ void outproj_kernel(const _Float16* __restrict__ oflat,
                               const _Float16* __restrict__ wout16,
                               const float* __restrict__ bout,
                               float* __restrict__ out)
{
  __shared__ _Float16 lds_b[2 * 64 * 32];
  const int blockM = blockIdx.x * 128;
  const int blockN = blockIdx.y * 64;
  const _Float16* A  = oflat  + (size_t)blockM * EMB;
  const _Float16* Bp = wout16 + (size_t)blockN * EMB;

  v8f acc[4] = {};
  gemm_core_128x64(A, EMB, Bp, EMB, EMB, lds_b, acc);

  const int lane = threadIdx.x & 31, wave = threadIdx.x >> 5;
  const int half16 = lane >> 4, l16 = lane & 15;
#pragma unroll
  for (int t = 0; t < 4; ++t) {
    const int n = blockN + t * 16 + l16;
    const float bv = bout[n];
#pragma unroll
    for (int g = 0; g < 8; ++g) {
      const int m = blockM + wave * 16 + g + half16 * 8;
      out[(size_t)m * EMB + n] = acc[t][g] + bv;
    }
  }
}

// ---------------------------------------------------------------------------
// attn_weights = mean over heads: [B][L][L] f32
// ---------------------------------------------------------------------------
__global__ void attnw_kernel(const _Float16* __restrict__ attn,
                             float* __restrict__ outw)
{
  const size_t total = (size_t)BATCH * L_SEQ * L_SEQ;
  const size_t idx = (size_t)blockIdx.x * blockDim.x + threadIdx.x;
  if (idx >= total) return;
  const int b = (int)(idx / ((size_t)L_SEQ * L_SEQ));
  const size_t r = idx % ((size_t)L_SEQ * L_SEQ);
  float s = 0.f;
#pragma unroll
  for (int h = 0; h < NHEAD; ++h)
    s += (float)attn[(size_t)(b * NHEAD + h) * L_SEQ * L_SEQ + r];
  outw[idx] = s * (1.0f / NHEAD);
}

// ---------------------------------------------------------------------------
extern "C" void kernel_launch(void* const* d_in, const int* in_sizes, int n_in,
                              void* d_out, int out_size, void* d_ws, size_t ws_size,
                              hipStream_t stream)
{
  const float* query = (const float*)d_in[0];
  const float* key_  = (const float*)d_in[1];
  const float* value = (const float*)d_in[2];
  const float* w_in  = (const float*)d_in[3];
  const float* b_in  = (const float*)d_in[4];
  const float* w_out = (const float*)d_in[5];
  const float* b_out = (const float*)d_in[6];

  char* ws = (char*)d_ws;
  size_t off = 0;
  auto carve = [&](size_t halfs) {
    _Float16* p = (_Float16*)(ws + off);
    off += halfs * sizeof(_Float16);
    return p;
  };
  _Float16* x16    = carve((size_t)3 * MROWS * EMB);       // q,k,v inputs f16
  _Float16* win16  = carve((size_t)3 * EMB * EMB);         // in_proj_weight f16
  _Float16* wout16 = carve((size_t)EMB * EMB);             // out_proj_weight f16
  _Float16* qh     = carve((size_t)BH * L_SEQ * HDIM);     // [bh][l][d] scaled
  _Float16* kh     = carve((size_t)BH * L_SEQ * HDIM);     // [bh][s][d]
  _Float16* vt     = carve((size_t)BH * HDIM * L_SEQ);     // [bh][d][s]
  _Float16* attn   = carve((size_t)BH * L_SEQ * L_SEQ);    // [bh][t][s]
  _Float16* oflat  = carve((size_t)MROWS * EMB);           // [L*B][E]

  float* out  = (float*)d_out;                              // [L,B,E]
  float* outw = out + (size_t)L_SEQ * BATCH * EMB;          // [B,L,L]

  const int nx = MROWS * EMB;  // 4M elems per q/k/v tensor
  cvt4_f32_f16_kernel<<<nx / 4 / 256, 256, 0, stream>>>(query, x16, nx / 4);
  cvt4_f32_f16_kernel<<<nx / 4 / 256, 256, 0, stream>>>(key_, x16 + (size_t)nx, nx / 4);
  cvt4_f32_f16_kernel<<<nx / 4 / 256, 256, 0, stream>>>(value, x16 + (size_t)2 * nx, nx / 4);
  cvt4_f32_f16_kernel<<<3 * EMB * EMB / 4 / 256, 256, 0, stream>>>(w_in, win16, 3 * EMB * EMB / 4);
  cvt4_f32_f16_kernel<<<EMB * EMB / 4 / 256, 256, 0, stream>>>(w_out, wout16, EMB * EMB / 4);

  qkv_proj_kernel<<<dim3(MROWS / 128, EMB / 64, 3), 256, 0, stream>>>(
      x16, win16, b_in, qh, kh, vt);
  scores_kernel<<<dim3(L_SEQ / 128, L_SEQ / 64, BH), 256, 0, stream>>>(qh, kh, attn);
  softmax_t_kernel<<<dim3(L_SEQ / 64, BH), 256, 0, stream>>>(attn);
  av_kernel<<<dim3(L_SEQ / 128, 1, BH), 256, 0, stream>>>(attn, vt, oflat);
  outproj_kernel<<<dim3(MROWS / 128, EMB / 64), 256, 0, stream>>>(oflat, wout16, b_out, out);
  attnw_kernel<<<(unsigned)(((size_t)BATCH * L_SEQ * L_SEQ) / 256), 256, 0, stream>>>(attn, outw);

  (void)in_sizes; (void)n_in; (void)out_size; (void)ws_size;
}
